// ResNet_77257871720901
// MI455X (gfx1250) — matrix operand order repaired and checked by reading, not compiled, not run on hardware
//
#include <hip/hip_runtime.h>
#include <hip/hip_bf16.h>

// ---------------------------------------------------------------------------
// Sparse 3D ResNet forward for MI455X (gfx1250).
// Convs = implicit GEMM on v_wmma_f32_16x16x32_bf16 (wave32, 16x16 tiles).
// bf16 activations / weights, f32 accumulate, fused mask * / residual +.
// All conv geometry is compile-time (3 template instantiations) so the
// per-wave prologue has no integer-division emulation.
// ---------------------------------------------------------------------------

#define S1v 95
#define S2v 47
#define BBv 2
#define C0v 8
#define C1v 16
#define NPTS 40000
#define EPSV 1e-4f

typedef __attribute__((ext_vector_type(16))) __bf16 v16bf;
typedef __attribute__((ext_vector_type(8)))  float  v8f;

union Frag { v16bf v; uint4 q[2]; };

__device__ __forceinline__ unsigned short f2bf(float f) {
  unsigned int u = __float_as_uint(f);
  u += 0x7FFFu + ((u >> 16) & 1u);   // round-to-nearest-even
  return (unsigned short)(u >> 16);
}

// ---------------- utility kernels -----------------------------------------

__global__ void k_zero(float* p, int n) {
  int i = blockIdx.x * blockDim.x + threadIdx.x;
  if (i < n) p[i] = 0.f;
}

__global__ void k_scatter(const int* __restrict__ coords,
                          const float* __restrict__ feats,
                          float* __restrict__ grid, float* __restrict__ mask) {
  int i = blockIdx.x * blockDim.x + threadIdx.x;
  if (i >= NPTS) return;
  int x = coords[i * 4 + 0];
  int y = coords[i * 4 + 1];
  int z = coords[i * 4 + 2];
  int b = coords[i * 4 + 3];
  int v = ((b * S1v + x) * S1v + y) * S1v + z;
  atomicAdd(&grid[v], feats[i]);
  mask[v] = 1.0f;
}

// first conv: 1 -> 8 channels, SAME, scalar FMA (K=27 tiny)
__global__ void k_conv0(const float* __restrict__ grid,
                        const float* __restrict__ mask,
                        const float* __restrict__ w0,
                        float* __restrict__ out, int nvox) {
  int v = blockIdx.x * blockDim.x + threadIdx.x;
  if (v >= nvox) return;
  int w = v % S1v; int t = v / S1v;
  int h = t % S1v; t /= S1v;
  int d = t % S1v; int b = t / S1v;
  float acc[C0v];
#pragma unroll
  for (int c = 0; c < C0v; ++c) acc[c] = 0.f;
  for (int tap = 0; tap < 27; ++tap) {
    int dz = tap / 9, rr = tap - dz * 9, dy = rr / 3, dx = rr - dy * 3;
    int di = d + dz - 1, hi = h + dy - 1, wi = w + dx - 1;
    if (di < 0 || di >= S1v || hi < 0 || hi >= S1v || wi < 0 || wi >= S1v) continue;
    float g = grid[((b * S1v + di) * S1v + hi) * S1v + wi];
#pragma unroll
    for (int c = 0; c < C0v; ++c) acc[c] += g * w0[tap * C0v + c];
  }
  float m = mask[v];
#pragma unroll
  for (int c = 0; c < C0v; ++c) out[(size_t)v * C0v + c] = acc[c] * m;
}

// per-channel sum / sumsq (stats[0..15]=sum, stats[16..31]=sumsq)
__global__ void k_stats(const float* __restrict__ x, float* __restrict__ stats,
                        int nvox, int C) {
  int c = blockIdx.y;
  __shared__ float ssum[256], ssq[256];
  float s = 0.f, q = 0.f;
  for (int i = blockIdx.x * blockDim.x + threadIdx.x; i < nvox;
       i += gridDim.x * blockDim.x) {
    float v = x[(size_t)i * C + c];
    s += v; q += v * v;
  }
  ssum[threadIdx.x] = s; ssq[threadIdx.x] = q;
  __syncthreads();
  for (int st = 128; st > 0; st >>= 1) {
    if (threadIdx.x < st) {
      ssum[threadIdx.x] += ssum[threadIdx.x + st];
      ssq[threadIdx.x]  += ssq[threadIdx.x + st];
    }
    __syncthreads();
  }
  if (threadIdx.x == 0) {
    atomicAdd(&stats[c], ssum[0]);
    atomicAdd(&stats[16 + c], ssq[0]);
  }
}

__global__ void k_count(const float* __restrict__ mask, float* __restrict__ cnt,
                        int nvox) {
  __shared__ float ssum[256];
  float s = 0.f;
  for (int i = blockIdx.x * blockDim.x + threadIdx.x; i < nvox;
       i += gridDim.x * blockDim.x) s += mask[i];
  ssum[threadIdx.x] = s;
  __syncthreads();
  for (int st = 128; st > 0; st >>= 1) {
    if (threadIdx.x < st) ssum[threadIdx.x] += ssum[threadIdx.x + st];
    __syncthreads();
  }
  if (threadIdx.x == 0) atomicAdd(cnt, ssum[0]);
}

// masked BN + ReLU + *mask; optional f32 and/or bf16 outputs.
// C is compile-time -> %C / /C become shift/mask-style strength-reduced ops.
template <int C>
__global__ void k_bnapply(const float* __restrict__ x,
                          const float* __restrict__ mask,
                          const float* __restrict__ stats,
                          const float* __restrict__ cnt,
                          const float* __restrict__ gamma,
                          const float* __restrict__ beta,
                          float* outF, unsigned short* outH, int nvox) {
  int i = blockIdx.x * blockDim.x + threadIdx.x;
  if (i >= nvox * C) return;
  int c = i % C;          // C = 8 or 16: bitwise AND
  int v = i / C;          // shift
  float n = cnt[0]; if (n < 1.f) n = 1.f;
  float mean = stats[c] / n;
  float var  = stats[16 + c] / n - mean * mean;
  float y = (x[i] - mean) * rsqrtf(var + EPSV) * gamma[c] + beta[c];
  y = fmaxf(y, 0.f);
  y *= mask[v];
  if (outF) outF[i] = y;
  if (outH) outH[i] = f2bf(y);
}

// pack weights (3,3,3,Cin,Cout) f32  ->  Wt[16][Kpad] bf16 (B^T layout, zero pad)
__global__ void k_prepw(const float* __restrict__ w, unsigned short* __restrict__ wt,
                        int Cin, int Cout, int Kpad) {
  int i = blockIdx.x * blockDim.x + threadIdx.x;
  if (i >= 16 * Kpad) return;
  int n = i / Kpad, k = i % Kpad;
  float val = 0.f;
  if (n < Cout && k < 27 * Cin) {
    int tap = k / Cin, ci = k % Cin;
    val = w[(tap * Cin + ci) * Cout + n];
  }
  wt[(size_t)n * Kpad + k] = f2bf(val);
}

// stride-2 VALID max-pool of the mask: 95^3 -> 47^3
__global__ void k_poolmask(const float* __restrict__ m1, float* __restrict__ m2,
                           int nvox2) {
  int v = blockIdx.x * blockDim.x + threadIdx.x;
  if (v >= nvox2) return;
  int w = v % S2v; int t = v / S2v;
  int h = t % S2v; t /= S2v;
  int d = t % S2v; int b = t / S2v;
  float mm = 0.f;
  for (int dz = 0; dz < 3; ++dz)
    for (int dy = 0; dy < 3; ++dy)
      for (int dx = 0; dx < 3; ++dx) {
        int di = 2 * d + dz, hi = 2 * h + dy, wi = 2 * w + dx;
        mm = fmaxf(mm, m1[((b * S1v + di) * S1v + hi) * S1v + wi]);
      }
  m2[v] = mm;
}

// ---------------- WMMA implicit-GEMM conv ---------------------------------
// One wave (one 32-thread block) -> 16 output voxels (consecutive W) x 16
// output channels. Grid = (tilesPerRow, SOUT, B*SOUT): no runtime division.
// A: 16 x 32 bf16 im2col rows (two b128 global loads per lane per chunk)
// B: 32 x 16 bf16 weights from Wt[c_out][KPAD] (two b128 loads per lane)
// D: f32 accum; epilogue fuses * mask (+ residual).
template <int CIN, int COUT, int SIN, int SOUT, int STRIDE, int POFS>
__global__ void k_conv_wmma(const unsigned short* __restrict__ act,
                            const unsigned short* __restrict__ wt,
                            const float* __restrict__ mask,
                            const float* __restrict__ res,
                            float* __restrict__ out, int hasRes) {
  constexpr int KPAD = (CIN == 8) ? 224 : 448;
  constexpr int NCH  = KPAD / 32;

  const int lane = threadIdx.x & 31;
  const int tw = blockIdx.x;
  const int hh = blockIdx.y;
  const int dd = blockIdx.z % SOUT;   // constant divisor -> mul/shift
  const int bb = blockIdx.z / SOUT;

  const int m  = lane & 15;       // A row index
  const int hf = lane >> 4;       // lane half
  const int wv = tw * 16 + m;     // this lane's A-row output W coordinate
  const bool rowOK = (wv < SOUT);
  const int n  = m;               // B / D channel column

  const int dbase = dd * STRIDE + POFS;
  const int hbase = hh * STRIDE + POFS;
  const int wbase = wv * STRIDE + POFS;

  // prefetch epilogue cachelines (global_prefetch_b8)
  const size_t voxbase = (size_t)(((bb * SOUT + dd) * SOUT + hh) * SOUT) + tw * 16;
  __builtin_prefetch(mask + voxbase, 0, 1);
  if (hasRes) __builtin_prefetch(res + voxbase * COUT, 0, 1);

  v8f acc = {0.f, 0.f, 0.f, 0.f, 0.f, 0.f, 0.f, 0.f};

#pragma unroll
  for (int ch = 0; ch < NCH; ++ch) {
    Frag a, b;
#pragma unroll
    for (int j = 0; j < 2; ++j) {
      // ISA 16-bit A layout: lane holds K = 8*hf..+7 (v0-3) and +16.. (v4-7)
      int k    = ch * 32 + 8 * hf + 16 * j;
      int tap  = k / CIN;             // compile-time per unrolled iteration
      int cofs = k % CIN;             // multiple of 8 -> 16B-aligned loads
      uint4 q = make_uint4(0u, 0u, 0u, 0u);
      if (rowOK && tap < 27) {
        int dz = tap / 9, rr = tap - dz * 9, dy = rr / 3, dx = rr - dy * 3;
        int di = dbase + dz, hi = hbase + dy, wi = wbase + dx;
        if (di >= 0 && di < SIN && hi >= 0 && hi < SIN && wi >= 0 && wi < SIN) {
          size_t addr =
              ((size_t)(((bb * SIN + di) * SIN + hi) * SIN + wi)) * CIN + cofs;
          q = *reinterpret_cast<const uint4*>(act + addr);
        }
      }
      a.q[j] = q;
    }
    {
      // ISA 16-bit B layout: lane n holds K = 8*hf..+7 and 16+8*hf..+7 of this chunk
      int kb = ch * 32 + 8 * hf;
      const uint4* wp = reinterpret_cast<const uint4*>(wt + (size_t)n * KPAD + kb);
      b.q[0] = wp[0];
      b.q[1] = wp[2];
    }
    acc = __builtin_amdgcn_wmma_f32_16x16x32_bf16(
        false, a.v, false, b.v, (short)0, acc, false, false);
  }

  if (n < COUT) {
#pragma unroll
    for (int r = 0; r < 8; ++r) {          // D: VGPR r <-> row r + 8*hf
      int wr = tw * 16 + r + 8 * hf;
      if (wr < SOUT) {
        size_t vox = voxbase + r + 8 * hf;
        float v = acc[r] * mask[vox];
        if (hasRes) v += res[vox * COUT + n];
        out[vox * COUT + n] = v;
      }
    }
  }
}

// ---------------- head -----------------------------------------------------

__global__ void k_pool(const float* __restrict__ x, const float* __restrict__ mask,
                       float* __restrict__ pooled) {
  int b = blockIdx.x / C1v, c = blockIdx.x % C1v;
  __shared__ float sm[256];
  const int per = S2v * S2v * S2v;
  float best = -3.0e38f;
  for (int i = threadIdx.x; i < per; i += blockDim.x) {
    int v = b * per + i;
    if (mask[v] > 0.f) best = fmaxf(best, x[(size_t)v * C1v + c]);
  }
  sm[threadIdx.x] = best;
  __syncthreads();
  for (int s = 128; s > 0; s >>= 1) {
    if (threadIdx.x < s) sm[threadIdx.x] = fmaxf(sm[threadIdx.x], sm[threadIdx.x + s]);
    __syncthreads();
  }
  if (threadIdx.x == 0) {
    float p = sm[0];
    if (p < -1.0e37f) p = 0.f;   // no voxel -> 0 (isfinite clamp)
    pooled[b * C1v + c] = p;
  }
}

__global__ void k_mlp(const float* __restrict__ pooled,
                      const float* __restrict__ l1w, const float* __restrict__ l1b,
                      const float* __restrict__ l2w, const float* __restrict__ l2b,
                      float* __restrict__ out) {
  __shared__ float h[BBv * 32];
  int tid = threadIdx.x;
  if (tid < BBv * 32) {
    int b = tid >> 5, j = tid & 31;
    float s = l1b[j];
    for (int c = 0; c < C1v; ++c) s += pooled[b * C1v + c] * l1w[j * C1v + c];
    h[tid] = s;
  }
  __syncthreads();
  if (tid < BBv) {
    float s = l2b[0];
    for (int j = 0; j < 32; ++j) s += h[tid * 32 + j] * l2w[j];
    out[tid] = s;
  }
}

// ---------------- host orchestration ---------------------------------------

extern "C" void kernel_launch(void* const* d_in, const int* in_sizes, int n_in,
                              void* d_out, int out_size, void* d_ws, size_t ws_size,
                              hipStream_t stream) {
  (void)in_sizes; (void)n_in; (void)out_size; (void)ws_size;

  const int NV1 = BBv * S1v * S1v * S1v;   // 1,714,750
  const int NV2 = BBv * S2v * S2v * S2v;   //   207,646

  const int*   coords = (const int*)d_in[0];
  const float* feats  = (const float*)d_in[1];
  const float* in[40];
  for (int i = 2; i < 40; ++i) in[i] = (const float*)d_in[i];

  // workspace carve-up
  char* base = (char*)d_ws;
  size_t off = 0;
  auto alloc = [&](size_t bytes) -> void* {
    void* p = base + off;
    off = (off + bytes + 255) & ~(size_t)255;
    return p;
  };
  float* grid  = (float*)alloc((size_t)NV1 * 4);
  float* mask1 = (float*)alloc((size_t)NV1 * 4);
  float* x8    = (float*)alloc((size_t)NV1 * C0v * 4);
  float* t8    = (float*)alloc((size_t)NV1 * C0v * 4);
  unsigned short* a8 = (unsigned short*)alloc((size_t)NV1 * C0v * 2);
  float* mask2 = (float*)alloc((size_t)NV2 * 4);
  float* x16   = (float*)alloc((size_t)NV2 * C1v * 4);
  float* t16   = (float*)alloc((size_t)NV2 * C1v * 4);
  unsigned short* a16 = (unsigned short*)alloc((size_t)NV2 * C1v * 2);
  float* y2    = (float*)alloc((size_t)NV2 * C1v * 4);
  float* stats = (float*)alloc(32 * 4);
  float* cnt1  = (float*)alloc(4);
  float* cnt2  = (float*)alloc(4);
  float* pooled = (float*)alloc(BBv * C1v * 4);
  unsigned short* wt[11];
  for (int i = 0; i < 11; ++i) wt[i] = (unsigned short*)alloc((size_t)16 * 448 * 2);

  auto Z = [&](float* p, int n) {
    k_zero<<<(n + 255) / 256, 256, 0, stream>>>(p, n);
  };
  auto prep = [&](unsigned short* dst, const float* src, int Cin, int Cout) {
    int Kpad = (Cin == 8) ? 224 : 448;
    int tot = 16 * Kpad;
    k_prepw<<<(tot + 255) / 256, 256, 0, stream>>>(src, dst, Cin, Cout, Kpad);
  };
  auto bn8 = [&](const float* xin, const float* mp, const float* cp,
                 const float* g, const float* b, float* outF,
                 unsigned short* outH, int nvox) {
    k_zero<<<1, 32, 0, stream>>>(stats, 32);
    dim3 gs(256, C0v);
    k_stats<<<gs, 256, 0, stream>>>(xin, stats, nvox, C0v);
    int total = nvox * C0v;
    k_bnapply<C0v><<<(total + 255) / 256, 256, 0, stream>>>(xin, mp, stats, cp,
                                                            g, b, outF, outH, nvox);
  };
  auto bn16 = [&](const float* xin, const float* mp, const float* cp,
                  const float* g, const float* b, float* outF,
                  unsigned short* outH, int nvox) {
    k_zero<<<1, 32, 0, stream>>>(stats, 32);
    dim3 gs(256, C1v);
    k_stats<<<gs, 256, 0, stream>>>(xin, stats, nvox, C1v);
    int total = nvox * C1v;
    k_bnapply<C1v><<<(total + 255) / 256, 256, 0, stream>>>(xin, mp, stats, cp,
                                                            g, b, outF, outH, nvox);
  };

  // conv launchers: grid = (tilesPerRow, SOUT, B*SOUT), one wave per block
  const dim3 gc1((S1v + 15) / 16, S1v, BBv * S1v);   // 8->8 SAME @95
  const dim3 gcd((S2v + 15) / 16, S2v, BBv * S2v);   // 95 -> 47 stride 2
  const dim3 gc2((S2v + 15) / 16, S2v, BBv * S2v);   // 16->16 SAME @47

  auto conv8_8 = [&](const unsigned short* ap, const unsigned short* wp,
                     const float* mp, const float* rp, float* op) {
    k_conv_wmma<8, 8, S1v, S1v, 1, -1>
        <<<gc1, 32, 0, stream>>>(ap, wp, mp, rp, op, rp != nullptr);
  };
  auto conv8_16s2 = [&](const unsigned short* ap, const unsigned short* wp,
                        const float* mp, const float* rp, float* op) {
    k_conv_wmma<8, 16, S1v, S2v, 2, 0>
        <<<gcd, 32, 0, stream>>>(ap, wp, mp, rp, op, rp != nullptr);
  };
  auto conv16_16 = [&](const unsigned short* ap, const unsigned short* wp,
                       const float* mp, const float* rp, float* op) {
    k_conv_wmma<16, 16, S2v, S2v, 1, -1>
        <<<gc2, 32, 0, stream>>>(ap, wp, mp, rp, op, rp != nullptr);
  };

  // ---- build grid + mask ----
  Z(grid, NV1); Z(mask1, NV1);
  k_scatter<<<(NPTS + 255) / 256, 256, 0, stream>>>(coords, feats, grid, mask1);

  // ---- pack conv weights for WMMA ----
  prep(wt[0],  in[7],  8,  8);   // ba0_w1
  prep(wt[1],  in[10], 8,  8);   // ba0_w2
  prep(wt[2],  in[13], 8,  8);   // ba1_w1
  prep(wt[3],  in[16], 8,  8);   // ba1_w2
  prep(wt[4],  in[31], 8,  16);  // dn_w1
  prep(wt[5],  in[35], 8,  16);  // dn_w3
  prep(wt[6],  in[34], 16, 16);  // dn_w2
  prep(wt[7],  in[19], 16, 16);  // bo0_w1
  prep(wt[8],  in[22], 16, 16);  // bo0_w2
  prep(wt[9],  in[25], 16, 16);  // bo1_w1
  prep(wt[10], in[28], 16, 16);  // bo1_w2

  // ---- mask counts ----
  k_zero<<<1, 1, 0, stream>>>(cnt1, 1);
  k_count<<<256, 256, 0, stream>>>(mask1, cnt1, NV1);

  // ---- stem: conv0 (1->8) + BN ----
  k_conv0<<<(NV1 + 255) / 256, 256, 0, stream>>>(grid, mask1, in[2], x8, NV1);
  bn8(x8, mask1, cnt1, in[3], in[4], x8, nullptr, NV1);          // x = bn0(x)

  // ---- ba0 ----
  bn8(x8, mask1, cnt1, in[5], in[6], nullptr, a8, NV1);
  conv8_8(a8, wt[0], mask1, nullptr, t8);
  bn8(t8, mask1, cnt1, in[8], in[9], nullptr, a8, NV1);
  conv8_8(a8, wt[1], mask1, x8, x8);                             // x += conv2

  // ---- ba1 ----
  bn8(x8, mask1, cnt1, in[11], in[12], nullptr, a8, NV1);
  conv8_8(a8, wt[2], mask1, nullptr, t8);
  bn8(t8, mask1, cnt1, in[14], in[15], nullptr, a8, NV1);
  conv8_8(a8, wt[3], mask1, x8, x8);

  // ---- downsample ----
  bn8(x8, mask1, cnt1, in[29], in[30], nullptr, a8, NV1);        // y = bn(x)
  k_poolmask<<<(NV2 + 255) / 256, 256, 0, stream>>>(mask1, mask2, NV2);
  k_zero<<<1, 1, 0, stream>>>(cnt2, 1);
  k_count<<<256, 256, 0, stream>>>(mask2, cnt2, NV2);
  conv8_16s2(a8, wt[4], mask2, nullptr, t16);                    // y1 = conv(y,w1,s2)
  conv8_16s2(a8, wt[5], mask2, nullptr, y2);                     // y2 = conv(y,w3,s2)
  bn16(t16, mask2, cnt2, in[32], in[33], nullptr, a16, NV2);
  conv16_16(a16, wt[6], mask2, y2, x16);                         // x = conv(..)+y2

  // ---- bo0 ----
  bn16(x16, mask2, cnt2, in[17], in[18], nullptr, a16, NV2);
  conv16_16(a16, wt[7], mask2, nullptr, t16);
  bn16(t16, mask2, cnt2, in[20], in[21], nullptr, a16, NV2);
  conv16_16(a16, wt[8], mask2, x16, x16);

  // ---- bo1 ----
  bn16(x16, mask2, cnt2, in[23], in[24], nullptr, a16, NV2);
  conv16_16(a16, wt[9], mask2, nullptr, t16);
  bn16(t16, mask2, cnt2, in[26], in[27], nullptr, a16, NV2);
  conv16_16(a16, wt[10], mask2, x16, x16);

  // ---- head: masked global max pool + MLP ----
  k_pool<<<BBv * C1v, 256, 0, stream>>>(x16, mask2, pooled);
  k_mlp<<<1, 64, 0, stream>>>(pooled, in[36], in[37], in[38], in[39],
                              (float*)d_out);
}
